// MultiHeadAttention_87548613361685
// MI455X (gfx1250) — matrix operand order repaired and checked by reading, not compiled, not run on hardware
//
#include <hip/hip_runtime.h>
#include <hip/hip_bf16.h>
#include <stdint.h>

// Problem constants (match reference)
#define SEQ   4096
#define EMB   768
#define HEADS 12
#define HD    64   // head dim; softmax scale = 1/sqrt(64) = 0.125

// ---------- gfx1250 feature probes (device pass only) ----------
#if defined(__gfx1250__) && __has_builtin(__builtin_amdgcn_global_load_async_to_lds_b128)
#define HAVE_ASYNC 1
#else
#define HAVE_ASYNC 0
#endif
#if defined(__gfx1250__) && __has_builtin(__builtin_amdgcn_tensor_load_to_lds)
#define HAVE_TDM 1
#else
#define HAVE_TDM 0
#endif

// ---------- types ----------
typedef float        v8f   __attribute__((ext_vector_type(8)));
typedef __bf16       v16bf __attribute__((ext_vector_type(16)));
typedef unsigned int u32x4 __attribute__((ext_vector_type(4)));
typedef int          i32x4 __attribute__((ext_vector_type(4)));
typedef int          i32x8 __attribute__((ext_vector_type(8)));

static __device__ __forceinline__ unsigned short f2bf(float f) {
  unsigned int u = __float_as_uint(f);
  u += 0x7fffu + ((u >> 16) & 1u);   // round-to-nearest-even
  return (unsigned short)(u >> 16);
}

struct Pair16 { u32x4 lo, hi; };
static __device__ __forceinline__ v16bf make_frag(u32x4 lo, u32x4 hi) {
  Pair16 p{lo, hi};
  return __builtin_bit_cast(v16bf, p);
}
static __device__ __forceinline__ u32x4 ld16(const unsigned short* p) {
  return *(const u32x4*)p;
}
static __device__ __forceinline__ v8f wmma_bf16(v16bf a, v16bf b, v8f c) {
  // D = A(16x32 bf16) * B(32x16 bf16) + C(16x16 f32)
  return __builtin_amdgcn_wmma_f32_16x16x32_bf16(false, a, false, b, (short)0, c,
                                                 false, false);
}

// ---------- async / TDM staging helpers ----------
#if HAVE_ASYNC || HAVE_TDM
typedef int v4i_gcc __attribute__((vector_size(16)));            // builtin's elem type
typedef __attribute__((address_space(1))) void*    gas_void;
typedef __attribute__((address_space(3))) void*    las_void;
typedef __attribute__((address_space(1))) v4i_gcc* gas_v4i;
typedef __attribute__((address_space(3))) v4i_gcc* las_v4i;
#endif

// 16B global -> LDS copy (async when available)
static __device__ __forceinline__ void stage16(const unsigned short* g,
                                               unsigned short* l) {
#if HAVE_ASYNC
  gas_void gv = (gas_void)(uintptr_t)g;   // global VA -> AS1
  las_void lv = (las_void)l;              // generic -> AS3
  __builtin_amdgcn_global_load_async_to_lds_b128((gas_v4i)gv, (las_v4i)lv, 0, 0);
#else
  *(u32x4*)l = *(const u32x4*)g;
#endif
}
static __device__ __forceinline__ void stage_wait() {
#if HAVE_ASYNC
#if __has_builtin(__builtin_amdgcn_s_wait_asynccnt)
  __builtin_amdgcn_s_wait_asynccnt(0);
#else
  asm volatile("s_wait_asynccnt 0" ::: "memory");
#endif
#endif
}

// ---------- elementwise conversion kernels ----------
__global__ __launch_bounds__(256) void cast_f32_bf16(const float* __restrict__ in,
                                                     unsigned short* __restrict__ out,
                                                     int n) {
  int i = blockIdx.x * 256 + threadIdx.x;
  if (i < n) out[i] = f2bf(in[i]);
}

// wT[n][k] = bf16(w[k][n])  (dim x dim)
__global__ __launch_bounds__(256) void transpose_f32_bf16(const float* __restrict__ w,
                                                          unsigned short* __restrict__ wT,
                                                          int dim) {
  int i = blockIdx.x * 256 + threadIdx.x;
  if (i < dim * dim) {
    int n = i / dim;
    int k = i - n * dim;
    wT[i] = f2bf(w[(size_t)k * dim + n]);
  }
}

// ---------- WMMA GEMM:  C[M=4096, N] = A[M,K] * BT[N,K]^T + bias ----------
// Block: 256 threads = 8 waves. Block tile 128(M) x 64(N), wave tile 16 x 64.
// Per k-step: A panel (128x32 bf16, 8KB) staged via async global->LDS loads,
// B panel (64x32 bf16, 4KB) staged via a single TDM 2-D descriptor (wave 0).
// mode 0: write bf16, head-split row-major  out[h][m][d]     (Q, K)
// mode 1: write bf16, head-split transposed out[h][d][m]     (V)
// mode 2: write f32 row-major               out[m][n]        (final projection)
__global__ __launch_bounds__(256) void gemm_kernel(
    const unsigned short* __restrict__ A,
    const unsigned short* __restrict__ BT,
    const float* __restrict__ bias,
    unsigned short* __restrict__ outb,
    float* __restrict__ outf,
    int K, int N, int mode, float scale)
{
  __shared__ __attribute__((aligned(16))) unsigned short ldsA[128 * 32]; // 128 M-rows x 32 K
  __shared__ __attribute__((aligned(16))) unsigned short ldsB[64 * 32];  // 64 N-rows x 32 K
  const int tid  = threadIdx.x;
  const int w    = tid >> 5;
  const int lane = tid & 31;
  const int nl   = lane & 15;
  const int half = lane >> 4;
  const int m0   = blockIdx.x * 128 + w * 16;
  const int n0   = blockIdx.y * 64;

  const v8f vzero = {0.f, 0.f, 0.f, 0.f, 0.f, 0.f, 0.f, 0.f};
  v8f acc[4] = {vzero, vzero, vzero, vzero};

  // A staging assignment: thread t copies 16B: row ai = t>>1, part ap = t&1
  const int ai = tid >> 1, ap = tid & 1;
  const unsigned short* agsrc = A + (size_t)(blockIdx.x * 128 + ai) * K + ap * 8;
  unsigned short*       aldst = ldsA + ai * 32 + ap * 8;
  // B staging fallback assignment: thread t copies 16B: row si = t>>2, part sp = t&3
  const int si = tid >> 2, sp = tid & 3;
  const unsigned short* bgsrc = BT + (size_t)(n0 + si) * K + sp * 8;
  unsigned short*       bldst = ldsB + si * 32 + sp * 8;
  (void)bgsrc; (void)bldst;

  // Per-wave A-fragment source in LDS: row = w*16 + nl
  const unsigned short* afrag = ldsA + (w * 16 + nl) * 32;

  for (int kb = 0; kb < K; kb += 32) {
    // ---- stage A panel: 256 threads x 16B async global->LDS ----
    stage16(agsrc + kb, aldst);

    // ---- stage B panel ----
#if HAVE_TDM
    if (w == 0) {
      const uintptr_t ga = (uintptr_t)(BT + (size_t)n0 * K + kb);
      const unsigned  la = (unsigned)(uintptr_t)(las_void)(void*)ldsB;
      u32x4 g0;
      g0.x = 1u;                                        // count=1, no gather
      g0.y = la;                                        // lds_addr (bytes)
      g0.z = (unsigned)(ga & 0xffffffffu);              // global_addr[31:0]
      g0.w = ((unsigned)(ga >> 32) & 0x01ffffffu)       // global_addr[56:32]
             | (2u << 30);                              // type = 2 ("image")
      i32x8 g1;
      g1[0] = 0x00010000;       // wg_mask=0, data_size=1 (2B), no pad/iterate
      g1[1] = (32 << 16);       // tensor_dim0 = 32 (lo16); atomic_barrier_addr=0
      g1[2] = (64 << 16);       // tensor_dim0 hi=0, tensor_dim1 = 64 (lo16)
      g1[3] = (32 << 16);       // tensor_dim1 hi=0, tile_dim0 = 32
      g1[4] = 64;               // tile_dim1 = 64, tile_dim2 = 0
      g1[5] = EMB;              // tensor_dim0_stride = K = 768 (lo32)
      g1[6] = 0;                // stride hi, tensor_dim1_stride lo
      g1[7] = 0;
      i32x4 gz = {0, 0, 0, 0};
#if __clang_major__ >= 23
      i32x8 gz8 = {0, 0, 0, 0, 0, 0, 0, 0};
      __builtin_amdgcn_tensor_load_to_lds(g0, g1, gz, gz, gz8, 0);
#else
      __builtin_amdgcn_tensor_load_to_lds(g0, g1, gz, gz, 0);
#endif
    }
#else
    stage16(bgsrc + kb, bldst);
#endif

    __builtin_prefetch(agsrc + kb + 32, 0, 3);   // near-cache prefetch next panels
#if !HAVE_TDM
    __builtin_prefetch(bgsrc + kb + 32, 0, 3);
#endif

    stage_wait();                                 // own ASYNCcnt drained
#if HAVE_TDM
    if (w == 0) __builtin_amdgcn_s_wait_tensorcnt(0);
#endif
    __syncthreads();                              // all panels visible to all waves

    // ---- A fragment from LDS: K chunks [half*8..+7] and [16+half*8..+7] ----
    v16bf a = make_frag(*(const u32x4*)(afrag + half * 8),
                        *(const u32x4*)(afrag + 16 + half * 8));

#pragma unroll
    for (int t = 0; t < 4; ++t) {
      // B fragment: lane N = t*16+nl; K chunks [half*8..+7] and [16+half*8..+7]
      const unsigned short* bp = ldsB + (t * 16 + nl) * 32 + half * 8;
      v16bf b = make_frag(*(const u32x4*)bp, *(const u32x4*)(bp + 16));
      acc[t] = wmma_bf16(a, b, acc[t]);
    }
    __syncthreads();
  }

#pragma unroll
  for (int t = 0; t < 4; ++t) {
    const int n = n0 + t * 16 + nl;
    const float bval = bias[n];
#pragma unroll
    for (int r = 0; r < 8; ++r) {
      const int m = m0 + half * 8 + r;     // C layout: lanes 0-15 rows 0-7, lanes 16-31 rows 8-15
      const float v = (acc[t][r] + bval) * scale;
      if (mode == 0)
        outb[((size_t)(n >> 6) * SEQ + m) * HD + (n & 63)] = f2bf(v);
      else if (mode == 1)
        outb[((size_t)(n >> 6) * HD + (n & 63)) * SEQ + m] = f2bf(v);
      else
        outf[(size_t)m * N + n] = v;
    }
  }
}

// ---------- Flash attention (causal, online softmax), WMMA throughout ----------
// Grid: (SEQ/128, HEADS). Block 256 = 8 independent waves, 16 queries per wave.
// Q pre-scaled by 1/sqrt(HD). Kb = [h][seq][d] bf16, Vt = [h][d][seq] bf16.
__global__ __launch_bounds__(256) void attn_kernel(
    const unsigned short* __restrict__ Qb,
    const unsigned short* __restrict__ Kb,
    const unsigned short* __restrict__ Vt,
    unsigned short* __restrict__ ctxb)
{
  __shared__ __attribute__((aligned(16))) unsigned short ldsP[8][16 * 32]; // per-wave P bounce
  const int tid  = threadIdx.x;
  const int w    = tid >> 5;
  const int lane = tid & 31;
  const int nl   = lane & 15;
  const int half = lane >> 4;
  const int h    = blockIdx.y;
  const int q0   = blockIdx.x * 128 + w * 16;   // wave's 16 queries: q0..q0+15

  // Q A-fragments, held in registers for the whole loop (d 0..31 and 32..63)
  const unsigned short* qrow = Qb + ((size_t)h * SEQ + q0 + nl) * HD;
  const v16bf aq0 = make_frag(ld16(qrow + half * 8),      ld16(qrow + 16 + half * 8));
  const v16bf aq1 = make_frag(ld16(qrow + 32 + half * 8), ld16(qrow + 48 + half * 8));

  const v8f vzero = {0.f, 0.f, 0.f, 0.f, 0.f, 0.f, 0.f, 0.f};
  v8f o[4] = {vzero, vzero, vzero, vzero};      // O accumulator, 16 x 64 f32
  float m8[8], l8[8];                           // row stats (rows half*8 + r)
#pragma unroll
  for (int r = 0; r < 8; ++r) { m8[r] = -__builtin_inff(); l8[r] = 0.f; }

  unsigned short* ldsW = &ldsP[w][0];
  const unsigned short* kbase = Kb + (size_t)h * SEQ * HD;

  for (int kb = 0; kb < q0 + 16; kb += 32) {    // causal: only key blocks touching keys <= q0+15
    __builtin_prefetch(kbase + (size_t)(kb + 32 + nl) * HD, 0, 3);

    // ---- S = Q * K^T : two 16x16 tiles (keys kb..kb+15 and kb+16..kb+31) ----
    v8f s0, s1;
    {
      const unsigned short* kr = kbase + (size_t)(kb + nl) * HD;
      v16bf b0 = make_frag(ld16(kr + half * 8),      ld16(kr + 16 + half * 8));
      v16bf b1 = make_frag(ld16(kr + 32 + half * 8), ld16(kr + 48 + half * 8));
      v8f st = vzero;
      st = wmma_bf16(aq0, b0, st);
      st = wmma_bf16(aq1, b1, st);
      s0 = st;
    }
    {
      const unsigned short* kr = kbase + (size_t)(kb + 16 + nl) * HD;
      v16bf b0 = make_frag(ld16(kr + half * 8),      ld16(kr + 16 + half * 8));
      v16bf b1 = make_frag(ld16(kr + 32 + half * 8), ld16(kr + 48 + half * 8));
      v8f st = vzero;
      st = wmma_bf16(aq0, b0, st);
      st = wmma_bf16(aq1, b1, st);
      s1 = st;
    }

    // ---- causal mask + online softmax (row reductions within 16-lane halves) ----
    const int col0 = kb + nl, col1 = kb + 16 + nl;
#pragma unroll
    for (int r = 0; r < 8; ++r) {
      const int row = q0 + half * 8 + r;
      float x0 = (col0 <= row) ? s0[r] : -__builtin_inff();
      float x1 = (col1 <= row) ? s1[r] : -__builtin_inff();
      float rm = fmaxf(x0, x1);
      rm = fmaxf(rm, __shfl_xor(rm, 1, 32));
      rm = fmaxf(rm, __shfl_xor(rm, 2, 32));
      rm = fmaxf(rm, __shfl_xor(rm, 4, 32));
      rm = fmaxf(rm, __shfl_xor(rm, 8, 32));
      float mn   = fmaxf(m8[r], rm);
      float corr = __expf(m8[r] - mn);
      float p0   = __expf(x0 - mn);
      float p1   = __expf(x1 - mn);
      float rs = p0 + p1;
      rs += __shfl_xor(rs, 1, 32);
      rs += __shfl_xor(rs, 2, 32);
      rs += __shfl_xor(rs, 4, 32);
      rs += __shfl_xor(rs, 8, 32);
      l8[r] = l8[r] * corr + rs;
      m8[r] = mn;
#pragma unroll
      for (int t = 0; t < 4; ++t) o[t][r] *= corr;
      // P tile (C layout) -> LDS row-major [16 rows][32 keys] bf16
      ldsW[(half * 8 + r) * 32 + nl]      = f2bf(p0);
      ldsW[(half * 8 + r) * 32 + 16 + nl] = f2bf(p1);
    }
    asm volatile("s_wait_dscnt 0" ::: "memory");   // P stores visible to the wave's loads

    // ---- P A-fragment (16 x 32 over keys) from LDS ----
    const unsigned short* pr = ldsW + nl * 32 + half * 8;
    v16bf pa = make_frag(*(const u32x4*)pr, *(const u32x4*)(pr + 16));

    // ---- O += P * V : V stored transposed [h][d][seq], contiguous along keys ----
#pragma unroll
    for (int t = 0; t < 4; ++t) {
      const unsigned short* vr =
          Vt + ((size_t)h * HD + t * 16 + nl) * SEQ + kb + half * 8;
      v16bf bvf = make_frag(ld16(vr), ld16(vr + 16));
      o[t] = wmma_bf16(pa, bvf, o[t]);
    }
  }

  // ---- finalize: ctx[row][h*64+d] = O / l  (bf16, feeds output projection) ----
#pragma unroll
  for (int r = 0; r < 8; ++r) {
    const int row = q0 + half * 8 + r;
    const float inv = 1.0f / l8[r];
#pragma unroll
    for (int t = 0; t < 4; ++t)
      ctxb[(size_t)row * EMB + h * HD + t * 16 + nl] = f2bf(o[t][r] * inv);
  }
}

// ---------- host-side orchestration ----------
extern "C" void kernel_launch(void* const* d_in, const int* in_sizes, int n_in,
                              void* d_out, int out_size, void* d_ws, size_t ws_size,
                              hipStream_t stream) {
  (void)in_sizes; (void)n_in; (void)out_size; (void)ws_size;

  const float* x  = (const float*)d_in[0];
  const float* wq = (const float*)d_in[1];
  const float* bq = (const float*)d_in[2];
  const float* wk = (const float*)d_in[3];
  const float* bk = (const float*)d_in[4];
  const float* wv = (const float*)d_in[5];
  const float* bv = (const float*)d_in[6];
  const float* wo = (const float*)d_in[7];
  const float* bo = (const float*)d_in[8];
  float* out = (float*)d_out;

  // Workspace carve-up (all bf16 buffers, ~34.5 MB total)
  char* ws = (char*)d_ws;
  const size_t XB  = (size_t)SEQ * EMB * 2;          // x in bf16
  const size_t WT  = (size_t)EMB * EMB * 2;          // one transposed weight
  const size_t QKV = (size_t)HEADS * SEQ * HD * 2;   // one head-split tensor
  unsigned short* xb  = (unsigned short*)(ws);
  unsigned short* wqT = (unsigned short*)(ws + XB);
  unsigned short* wkT = (unsigned short*)(ws + XB + 1 * WT);
  unsigned short* wvT = (unsigned short*)(ws + XB + 2 * WT);
  unsigned short* woT = (unsigned short*)(ws + XB + 3 * WT);
  unsigned short* Qb  = (unsigned short*)(ws + XB + 4 * WT);
  unsigned short* Kb  = (unsigned short*)(ws + XB + 4 * WT + 1 * QKV);
  unsigned short* Vt  = (unsigned short*)(ws + XB + 4 * WT + 2 * QKV);
  unsigned short* ctx = (unsigned short*)(ws + XB + 4 * WT + 3 * QKV);

  // 1) precision conversion (x -> bf16, weights -> transposed bf16)
  cast_f32_bf16<<<(SEQ * EMB) / 256, 256, 0, stream>>>(x, xb, SEQ * EMB);
  transpose_f32_bf16<<<(EMB * EMB) / 256, 256, 0, stream>>>(wq, wqT, EMB);
  transpose_f32_bf16<<<(EMB * EMB) / 256, 256, 0, stream>>>(wk, wkT, EMB);
  transpose_f32_bf16<<<(EMB * EMB) / 256, 256, 0, stream>>>(wv, wvT, EMB);
  transpose_f32_bf16<<<(EMB * EMB) / 256, 256, 0, stream>>>(wo, woT, EMB);

  // 2) QKV projections (Q pre-scaled by 1/sqrt(HD); V stored transposed)
  dim3 gg(SEQ / 128, EMB / 64);
  gemm_kernel<<<gg, 256, 0, stream>>>(xb, wqT, bq, Qb, nullptr, EMB, EMB, 0, 0.125f);
  gemm_kernel<<<gg, 256, 0, stream>>>(xb, wkT, bk, Kb, nullptr, EMB, EMB, 0, 1.0f);
  gemm_kernel<<<gg, 256, 0, stream>>>(xb, wvT, bv, Vt, nullptr, EMB, EMB, 1, 1.0f);

  // 3) causal flash attention
  attn_kernel<<<dim3(SEQ / 128, HEADS), 256, 0, stream>>>(Qb, Kb, Vt, ctx);

  // 4) output projection -> f32 d_out
  gemm_kernel<<<gg, 256, 0, stream>>>(ctx, woT, bo, nullptr, out, EMB, EMB, 2, 1.0f);
}
